// PRGAT_34651796144594
// MI455X (gfx1250) — matrix operand order repaired and checked by reading, not compiled, not run on hardware
//
#include <hip/hip_runtime.h>
#include <hip/hip_bf16.h>

typedef __attribute__((ext_vector_type(16))) __bf16 v16bf;
typedef __attribute__((ext_vector_type(8)))  float  v8f;

// ---------------------------------------------------------------------------
// Tall-skinny GEMM: C[M,Kout] = A[M,Kin](f32) * Wf(bf16, fragment-packed)
// f32 accumulate via v_wmma_f32_16x16x32_bf16.
// Block = 256 threads = 8 waves. Block tile = 64 rows x 64 cols.
// Wave w: rows rowBase+(w>>1)*16 ; cols colBase+(w&1)*32 (two 16-wide subtiles).
// Wf layout: [Kin/32][Kout/16][lane 0..31][16 bf16]  (one v16bf per lane)
// ---------------------------------------------------------------------------
__global__ __launch_bounds__(256)
void gemm_bf16_wmma(const float* __restrict__ A, const __bf16* __restrict__ Wf,
                    float* __restrict__ C, int M, int Kin, int Kout) {
  // 64 rows x 32 k, padded to 40 bf16 (80B row stride; all tile accesses 16B-aligned)
  __shared__ __bf16 As[64][40];
  const int tid  = threadIdx.x;
  const int wave = tid >> 5;
  const int lane = tid & 31;
  const int half = lane >> 4;
  const int l16  = lane & 15;
  const int rowBase = blockIdx.x * 64;
  const int colBase = blockIdx.y * 64 + (wave & 1) * 32;
  const int waveRow = (wave >> 1) * 16;
  const int nTiles  = Kout >> 4;

  // A staging: each thread owns 8 consecutive k in one row (64*32/256 == 8)
  const int ar = tid >> 2;              // 0..63
  const int ac = (tid & 3) * 8;         // 0,8,16,24
  int gr = rowBase + ar; if (gr >= M) gr = M - 1;   // clamp; stores guarded
  const float* aptr = A + (long)gr * Kin + ac;

  // wave-uniform activity flags -> scalar branches (EXEC stays full at WMMA)
  const bool act0 = __builtin_amdgcn_readfirstlane((int)(colBase      < Kout)) != 0;
  const bool act1 = __builtin_amdgcn_readfirstlane((int)(colBase + 16 < Kout)) != 0;

  v8f acc0 = {}; v8f acc1 = {};

  // 2-stage software pipeline on the A slab
  float4 f0 = *(const float4*)(aptr);
  float4 f1 = *(const float4*)(aptr + 4);

  for (int kb = 0; kb < Kin; kb += 32) {
    __syncthreads();
    union { __bf16 h[8]; uint4 u; } cv;
    cv.h[0] = (__bf16)f0.x; cv.h[1] = (__bf16)f0.y;
    cv.h[2] = (__bf16)f0.z; cv.h[3] = (__bf16)f0.w;
    cv.h[4] = (__bf16)f1.x; cv.h[5] = (__bf16)f1.y;
    cv.h[6] = (__bf16)f1.z; cv.h[7] = (__bf16)f1.w;
    *(uint4*)&As[ar][ac] = cv.u;
    __syncthreads();

    // issue next slab's global loads now; they complete during the WMMAs
    if (kb + 32 < Kin) {
      f0 = *(const float4*)(aptr + kb + 32);
      f1 = *(const float4*)(aptr + kb + 36);
    }

    // A fragment: contiguous k = half*16 + s (matches packed B), two 16B runs per lane in LDS
    const int m = waveRow + l16;
    union { uint4 u[2]; v16bf v; } au;
    au.u[0] = *(const uint4*)&As[m][half * 16];
    au.u[1] = *(const uint4*)&As[m][half * 16 + 8];

    // both B fragments first (one wait covers both), then back-to-back WMMAs
    const long fragBase = (long)(kb >> 5) * nTiles * 512;
    v16bf bv0 = {}; v16bf bv1 = {};
    if (act0) bv0 = *(const v16bf*)(Wf + fragBase +
                                    ((long)(colBase >> 4) * 32 + lane) * 16);
    if (act1) bv1 = *(const v16bf*)(Wf + fragBase +
                                    ((long)((colBase + 16) >> 4) * 32 + lane) * 16);
    if (act0) acc0 = __builtin_amdgcn_wmma_f32_16x16x32_bf16(false, au.v, false, bv0,
                                                             (short)0, acc0, false, false);
    if (act1) acc1 = __builtin_amdgcn_wmma_f32_16x16x32_bf16(false, au.v, false, bv1,
                                                             (short)0, acc1, false, false);
  }

  // C/D layout: VGPR r, lane l -> row = r + (l>>4)*8, col = l&15
#pragma unroll
  for (int r = 0; r < 8; ++r) {
    int row = rowBase + waveRow + half * 8 + r;
    if (row < M) {
      if (act0) C[(long)row * Kout + colBase + l16]      = acc0[r];
      if (act1) C[(long)row * Kout + colBase + 16 + l16] = acc1[r];
    }
  }
}

// ---------------------------------------------------------------------------
// Weight packing into WMMA fragment order: out[kbIdx][tile][lane][j] =
//   W[kbIdx*32 + (lane>>4)*16 + j][tile*16 + (lane&15)]
// ---------------------------------------------------------------------------
__global__ void pack_frag(const float* __restrict__ W, __bf16* __restrict__ out,
                          int Kin, int Kout) {
  int idx = blockIdx.x * 256 + threadIdx.x;
  if (idx >= Kin * Kout) return;
  int j    = idx & 15;
  int lane = (idx >> 4) & 31;
  int rest = idx >> 9;
  int nT   = Kout >> 4;
  int tile = rest % nT;
  int kbI  = rest / nT;
  int k = kbI * 32 + (lane >> 4) * 16 + j;
  int n = tile * 16 + (lane & 15);
  out[idx] = (__bf16)W[(long)k * Kout + n];
}

// RGCN pack: virtual W[F,320] = [bases(b=0..3, col b*64+o) || root(col 256+o)]
__global__ void pack_rgcn_frag(const float* __restrict__ bases,
                               const float* __restrict__ root,
                               __bf16* __restrict__ out, int F) {
  const int Kout = 320;
  int idx = blockIdx.x * 256 + threadIdx.x;
  if (idx >= F * Kout) return;
  int j    = idx & 15;
  int lane = (idx >> 4) & 31;
  int rest = idx >> 9;
  int nT   = Kout >> 4;            // 20
  int tile = rest % nT;
  int kbI  = rest / nT;
  int k = kbI * 32 + (lane >> 4) * 16 + j;     // feature index f
  int n = tile * 16 + (lane & 15);             // output col
  float v;
  if (n < 256) { int b = n >> 6, o = n & 63; v = bases[((long)b * F + k) * 64 + o]; }
  else         { v = root[(long)k * 64 + (n - 256)]; }
  out[idx] = (__bf16)v;
}

// ---------------------------------------------------------------------------
// Utility
// ---------------------------------------------------------------------------
__global__ void zero_f32(float* __restrict__ p, long n) {
  long i = (long)blockIdx.x * 256 + threadIdx.x;
  if (i < n) p[i] = 0.0f;
}

// ---------------------------------------------------------------------------
// RGCN sparse kernels
// ---------------------------------------------------------------------------
__global__ void rgcn_count(const int* __restrict__ dst, const int* __restrict__ et,
                           float* __restrict__ cnt, int E) {
  int e = blockIdx.x * 256 + threadIdx.x;
  if (e < E) atomicAdd(&cnt[(long)dst[e] * 8 + et[e]], 1.0f);
}

// one wave per edge; lanes cover the 64 output features (2 each)
__global__ __launch_bounds__(256)
void rgcn_edge(const int* __restrict__ src, const int* __restrict__ dst,
               const int* __restrict__ et, const float* __restrict__ comp,
               const float* __restrict__ cnt, const float* __restrict__ xb,
               float* __restrict__ agg, int E) {
  int wv   = (int)(((long)blockIdx.x * 256 + threadIdx.x) >> 5);
  int lane = threadIdx.x & 31;
  if (wv >= E) return;
  int s = src[wv], d = dst[wv], t = et[wv];
  float c0 = comp[t * 4 + 0], c1 = comp[t * 4 + 1];
  float c2 = comp[t * 4 + 2], c3 = comp[t * 4 + 3];
  float norm = 1.0f / fmaxf(cnt[(long)d * 8 + t], 1.0f);
  const float* xs = xb + (long)s * 320;
#pragma unroll
  for (int q = 0; q < 2; ++q) {
    int o = lane + q * 32;
    float m = c0 * xs[o] + c1 * xs[64 + o] + c2 * xs[128 + o] + c3 * xs[192 + o];
    atomicAdd(&agg[(long)d * 64 + o], m * norm);
  }
}

__global__ void rgcn_finalize(const float* __restrict__ agg, const float* __restrict__ proj,
                              const float* __restrict__ bias, float* __restrict__ h, int N) {
  long i = (long)blockIdx.x * 256 + threadIdx.x;
  if (i >= (long)N * 64) return;
  int n = (int)(i >> 6), o = (int)(i & 63);
  float v = agg[i] + proj[(long)n * 320 + 256 + o] + bias[o];
  h[i] = fmaxf(v, 0.0f);
}

// ---------------------------------------------------------------------------
// GAT sparse kernels (self-loops via extended edge index e >= E)
// ---------------------------------------------------------------------------
__global__ void gat_coeff(const float* __restrict__ xh, const float* __restrict__ asrc,
                          const float* __restrict__ adst, float* __restrict__ alS,
                          float* __restrict__ alD, int N, int H, int Cc) {
  int i = blockIdx.x * 256 + threadIdx.x;
  if (i >= N * H) return;
  int n = i / H, h = i % H;
  const float* xp = xh + ((long)n * H + h) * Cc;
  float ss = 0.f, sd = 0.f;
  for (int c = 0; c < Cc; ++c) { ss += xp[c] * asrc[h * Cc + c]; sd += xp[c] * adst[h * Cc + c]; }
  alS[i] = ss; alD[i] = sd;
}

__global__ void gat_den(const int* __restrict__ src, const int* __restrict__ dst,
                        const float* __restrict__ alS, const float* __restrict__ alD,
                        float* __restrict__ den, int E, int N, int H) {
  int e = blockIdx.x * 256 + threadIdx.x;
  if (e >= E + N) return;
  int s, d;
  if (e < E) { s = src[e]; d = dst[e]; } else { s = e - E; d = s; }
  for (int h = 0; h < H; ++h) {
    float v = alS[s * H + h] + alD[d * H + h];
    v = v > 0.f ? v : 0.2f * v;
    atomicAdd(&den[d * H + h], __expf(v));
  }
}

// one wave per extended edge; lanes stride over H*Cc features
__global__ __launch_bounds__(256)
void gat_agg(const int* __restrict__ src, const int* __restrict__ dst,
             const float* __restrict__ alS, const float* __restrict__ alD,
             const float* __restrict__ den, const float* __restrict__ xh,
             float* __restrict__ out, int E, int N, int H, int Cc) {
  int wv   = (int)(((long)blockIdx.x * 256 + threadIdx.x) >> 5);
  int lane = threadIdx.x & 31;
  if (wv >= E + N) return;
  int s, d;
  if (wv < E) { s = src[wv]; d = dst[wv]; } else { s = wv - E; d = s; }
  int HC = H * Cc;
  for (int idx = lane; idx < HC; idx += 32) {
    int h = idx / Cc;
    float v = alS[s * H + h] + alD[d * H + h];
    v = v > 0.f ? v : 0.2f * v;
    float alpha = __expf(v) / den[d * H + h];
    atomicAdd(&out[(long)d * HC + idx], alpha * xh[(long)s * HC + idx]);
  }
}

__global__ void gat_finalize(const float* __restrict__ acc, const float* __restrict__ bias,
                             float* __restrict__ out, long n, int HC, int do_relu) {
  long i = (long)blockIdx.x * 256 + threadIdx.x;
  if (i >= n) return;
  float v = acc[i] + bias[(int)(i % HC)];
  out[i] = do_relu ? fmaxf(v, 0.0f) : v;
}

__global__ void final_tanh(const float* __restrict__ h, float* __restrict__ out, int N) {
  int n = blockIdx.x * 256 + threadIdx.x;
  if (n >= N) return;
  float s = 0.f;
#pragma unroll
  for (int j = 0; j < 16; ++j) s += h[(long)n * 16 + j];
  out[n] = tanhf(s * (1.0f / 16.0f));
}

// ---------------------------------------------------------------------------
// Host orchestration
// ---------------------------------------------------------------------------
static inline int cdiv(long a, long b) { return (int)((a + b - 1) / b); }

extern "C" void kernel_launch(void* const* d_in, const int* in_sizes, int n_in,
                              void* d_out, int out_size, void* d_ws, size_t ws_size,
                              hipStream_t stream) {
  const int N = in_sizes[0] / 32;      // nodes
  const int E = in_sizes[2];           // edges

  const float* x     = (const float*)d_in[0];
  const int*   eidx  = (const int*)d_in[1];
  const int*   etype = (const int*)d_in[2];
  const int*   esrc  = eidx;
  const int*   edst  = eidx + E;
  const float* r0_bases = (const float*)d_in[3];
  const float* r0_comp  = (const float*)d_in[4];
  const float* r0_root  = (const float*)d_in[5];
  const float* r0_bias  = (const float*)d_in[6];
  const float* r1_bases = (const float*)d_in[7];
  const float* r1_comp  = (const float*)d_in[8];
  const float* r1_root  = (const float*)d_in[9];
  const float* r1_bias  = (const float*)d_in[10];
  const float* g0_w = (const float*)d_in[11], *g0_as = (const float*)d_in[12];
  const float* g0_ad = (const float*)d_in[13], *g0_b = (const float*)d_in[14];
  const float* g1_w = (const float*)d_in[15], *g1_as = (const float*)d_in[16];
  const float* g1_ad = (const float*)d_in[17], *g1_b = (const float*)d_in[18];
  const float* go_w = (const float*)d_in[19], *go_as = (const float*)d_in[20];
  const float* go_ad = (const float*)d_in[21], *go_b = (const float*)d_in[22];
  float* outp = (float*)d_out;

  // workspace partition (floats)
  float* ws = (float*)d_ws;
  long off = 0;
  float* bufA = ws + off; off += (long)N * 320;   // GEMM outputs (xb||root proj, xh)
  float* bufB = ws + off; off += (long)N * 128;   // current hidden
  float* bufC = ws + off; off += (long)N * 128;   // aggregation accumulator
  float* cnt  = ws + off; off += (long)N * 8;     // per (dst, rel) counts
  float* alS  = ws + off; off += (long)N * 2;
  float* alD  = ws + off; off += (long)N * 2;
  float* den  = ws + off; off += (long)N * 2;
  __bf16* wcat0 = (__bf16*)(ws + off);
  __bf16* wcat1 = wcat0 + 32 * 320;
  __bf16* wg0   = wcat1 + 64 * 320;
  __bf16* wg1   = wg0 + 64 * 128;
  __bf16* wgo   = wg1 + 128 * 128;

  const int TB = 256;
  const int E2 = E + N;
  dim3 blk(TB);

  // --- weight prep: fp32 -> bf16, WMMA fragment order ---
  pack_rgcn_frag<<<cdiv(32 * 320, TB), blk, 0, stream>>>(r0_bases, r0_root, wcat0, 32);
  pack_rgcn_frag<<<cdiv(64 * 320, TB), blk, 0, stream>>>(r1_bases, r1_root, wcat1, 64);
  pack_frag<<<cdiv(64 * 128, TB), blk, 0, stream>>>(g0_w, wg0, 64, 128);
  pack_frag<<<cdiv(128 * 128, TB), blk, 0, stream>>>(g1_w, wg1, 128, 128);
  pack_frag<<<cdiv(128 * 16, TB), blk, 0, stream>>>(go_w, wgo, 128, 16);

  // --- per-(dst,rel) degree counts (shared by both RGCN layers) ---
  zero_f32<<<cdiv((long)N * 8, TB), blk, 0, stream>>>(cnt, (long)N * 8);
  rgcn_count<<<cdiv(E, TB), blk, 0, stream>>>(edst, etype, cnt, E);

  const int edgeWaveBlocks  = cdiv((long)E * 32, TB);
  const int edgeWaveBlocks2 = cdiv((long)E2 * 32, TB);
  dim3 ggrid(cdiv(N, 64), 0, 1);

  // --- RGCN layer 0: x[N,32] -> h[N,64] ---
  ggrid.y = cdiv(320, 64);
  gemm_bf16_wmma<<<ggrid, blk, 0, stream>>>(x, wcat0, bufA, N, 32, 320);
  zero_f32<<<cdiv((long)N * 64, TB), blk, 0, stream>>>(bufC, (long)N * 64);
  rgcn_edge<<<edgeWaveBlocks, blk, 0, stream>>>(esrc, edst, etype, r0_comp, cnt, bufA, bufC, E);
  rgcn_finalize<<<cdiv((long)N * 64, TB), blk, 0, stream>>>(bufC, bufA, r0_bias, bufB, N);

  // --- RGCN layer 1: h[N,64] -> h[N,64] ---
  gemm_bf16_wmma<<<ggrid, blk, 0, stream>>>(bufB, wcat1, bufA, N, 64, 320);
  zero_f32<<<cdiv((long)N * 64, TB), blk, 0, stream>>>(bufC, (long)N * 64);
  rgcn_edge<<<edgeWaveBlocks, blk, 0, stream>>>(esrc, edst, etype, r1_comp, cnt, bufA, bufC, E);
  rgcn_finalize<<<cdiv((long)N * 64, TB), blk, 0, stream>>>(bufC, bufA, r1_bias, bufB, N);

  // --- GAT layer 0: h[N,64] -> h[N,128] (H=2, C=64, relu) ---
  ggrid.y = cdiv(128, 64);
  gemm_bf16_wmma<<<ggrid, blk, 0, stream>>>(bufB, wg0, bufA, N, 64, 128);
  gat_coeff<<<cdiv(N * 2, TB), blk, 0, stream>>>(bufA, g0_as, g0_ad, alS, alD, N, 2, 64);
  zero_f32<<<cdiv((long)N * 2, TB), blk, 0, stream>>>(den, (long)N * 2);
  gat_den<<<cdiv(E2, TB), blk, 0, stream>>>(esrc, edst, alS, alD, den, E, N, 2);
  zero_f32<<<cdiv((long)N * 128, TB), blk, 0, stream>>>(bufC, (long)N * 128);
  gat_agg<<<edgeWaveBlocks2, blk, 0, stream>>>(esrc, edst, alS, alD, den, bufA, bufC, E, N, 2, 64);
  gat_finalize<<<cdiv((long)N * 128, TB), blk, 0, stream>>>(bufC, g0_b, bufB, (long)N * 128, 128, 1);

  // --- GAT layer 1: h[N,128] -> h[N,128] (relu) ---
  gemm_bf16_wmma<<<ggrid, blk, 0, stream>>>(bufB, wg1, bufA, N, 128, 128);
  gat_coeff<<<cdiv(N * 2, TB), blk, 0, stream>>>(bufA, g1_as, g1_ad, alS, alD, N, 2, 64);
  zero_f32<<<cdiv((long)N * 2, TB), blk, 0, stream>>>(den, (long)N * 2);
  gat_den<<<cdiv(E2, TB), blk, 0, stream>>>(esrc, edst, alS, alD, den, E, N, 2);
  zero_f32<<<cdiv((long)N * 128, TB), blk, 0, stream>>>(bufC, (long)N * 128);
  gat_agg<<<edgeWaveBlocks2, blk, 0, stream>>>(esrc, edst, alS, alD, den, bufA, bufC, E, N, 2, 64);
  gat_finalize<<<cdiv((long)N * 128, TB), blk, 0, stream>>>(bufC, g1_b, bufB, (long)N * 128, 128, 1);

  // --- GAT output layer: h[N,128] -> h[N,16] (H=2, C=8, no relu) ---
  ggrid.y = cdiv(16, 64);
  gemm_bf16_wmma<<<ggrid, blk, 0, stream>>>(bufB, wgo, bufA, N, 128, 16);
  gat_coeff<<<cdiv(N * 2, TB), blk, 0, stream>>>(bufA, go_as, go_ad, alS, alD, N, 2, 8);
  zero_f32<<<cdiv((long)N * 2, TB), blk, 0, stream>>>(den, (long)N * 2);
  gat_den<<<cdiv(E2, TB), blk, 0, stream>>>(esrc, edst, alS, alD, den, E, N, 2);
  zero_f32<<<cdiv((long)N * 16, TB), blk, 0, stream>>>(bufC, (long)N * 16);
  gat_agg<<<edgeWaveBlocks2, blk, 0, stream>>>(esrc, edst, alS, alD, den, bufA, bufC, E, N, 2, 8);
  gat_finalize<<<cdiv((long)N * 16, TB), blk, 0, stream>>>(bufC, go_b, bufB, (long)N * 16, 16, 0);

  // --- logits = tanh(mean over 16 features) ---
  final_tanh<<<cdiv(N, TB), blk, 0, stream>>>(bufB, outp, N);

  (void)n_in; (void)out_size; (void)ws_size;
}
